// Sp_GIN_8383776162609
// MI455X (gfx1250) — matrix operand order repaired and checked by reading, not compiled, not run on hardware
//
#include <hip/hip_runtime.h>
#include <hip/hip_bf16.h>

// ---------------------------------------------------------------------------
// Types for CDNA5 WMMA (gfx1250, wave32)
// ---------------------------------------------------------------------------
typedef __attribute__((ext_vector_type(16))) __bf16 v16bf;
typedef __attribute__((ext_vector_type(8)))  float  v8f;

#define HID 256
#define N_NODES 50000
#define N_EDGES 800000
#define IN_DIM 162
#define K1PAD 192                 // IN_DIM padded to multiple of 32
#define MPAD 50048                // N_NODES padded to multiple of 64

// ---------------------------------------------------------------------------
// Zero a small float region (reduction accumulators)
// ---------------------------------------------------------------------------
__global__ void zero_kernel(float* p, int n) {
    int i = blockIdx.x * blockDim.x + threadIdx.x;
    if (i < n) p[i] = 0.0f;
}

// Copy feat [M x IN_DIM] -> dst [.. x K1PAD], zero the pad columns.
__global__ void pad_copy_feat(const float* __restrict__ src,
                              float* __restrict__ dst, int M) {
    int i = blockIdx.x * blockDim.x + threadIdx.x;
    const int n = M * K1PAD;
    if (i >= n) return;
    const int r = i / K1PAD;
    const int c = i - r * K1PAD;
    dst[i] = (c < IN_DIM) ? src[r * IN_DIM + c] : 0.0f;
}

// Pad weight [IN_DIM x HID] -> [K1PAD x HID], zero pad rows. Same row stride.
__global__ void pad_weight(const float* __restrict__ w,
                           float* __restrict__ wp) {
    int i = blockIdx.x * blockDim.x + threadIdx.x;
    if (i >= K1PAD * HID) return;
    const int r = i / HID;
    wp[i] = (r < IN_DIM) ? w[i] : 0.0f;
}

// ---------------------------------------------------------------------------
// Edge scatter:  acc[dst] += x[src]   (acc pre-seeded with x => x + agg)
// Coalesced row reads + coalesced global_atomic_add_f32. Separate strides.
// ---------------------------------------------------------------------------
__global__ void scatter_add_kernel(const float* __restrict__ X, int ldx,
                                   const long long* __restrict__ ei,
                                   float* __restrict__ acc, int lda,
                                   int D, int nE) {
    for (int e = blockIdx.x; e < nE; e += gridDim.x) {
        long long s = ei[e];
        long long d = ei[(long long)nE + e];
        const float* xs = X + s * (long long)ldx;
        float*       ad = acc + d * (long long)lda;
        for (int f = threadIdx.x; f < D; f += blockDim.x)
            atomicAdd(&ad[f], xs[f]);
    }
}

// ---------------------------------------------------------------------------
// WMMA GEMM + bias + ReLU:  C[M x HID] = relu(A[M x K] @ W[K x HID] + b)
//
// Each wave computes a 64(M) x 16(N) strip: one B fragment per K-step reused
// by 4 v_wmma_f32_16x16x32_bf16 (B loads amortized 4x). A fragments are four
// global_load_b128 each (K-contiguous per the ISA layout). K must be a
// multiple of 32 (inputs padded); A rows up to MPAD must be readable
// (garbage pad rows OK: their outputs are store-guarded).
//
// Fragment layouts per CDNA5 ISA 7.12.2 (wave32):
//  A (16x32 bf16): lane L -> row M = L&15, half = L>>4.
//     elements 0-7:  K = k0 + half*8 + e ; elements 8-15: K = k0+16+half*8+(e-8)
//  B (32x16 bf16): lane L -> col N = L&15; element e -> same K as A element e
//  C/D (16x16 f32): VGPR r -> row M = r + 8*half, col N = L&15
// ---------------------------------------------------------------------------
__global__ void gemm_bias_relu_wmma(const float* __restrict__ A, int lda,
                                    const float* __restrict__ W,
                                    const float* __restrict__ bias,
                                    float* __restrict__ C,
                                    int M, int K) {
    const int lane = threadIdx.x & 31;
    const int wave = threadIdx.x >> 5;
    const int tile = blockIdx.x * 8 + wave;      // 8 waves / block
    const int ntN  = HID / 16;                   // 16 N-tiles
    const int mt   = tile / ntN;                 // 64-row supertile
    const int nt   = tile % ntN;

    const int m0   = mt * 64;
    const int n0   = nt * 16;
    const int half = lane >> 4;
    const int mrow = m0 + (lane & 15);           // base row (subtile 0)
    const int ncol = n0 + (lane & 15);

    const float* __restrict__ Arow = A + (size_t)mrow * lda;
    const float* __restrict__ Wcol = W + ncol;

    v8f acc[4] = {{}, {}, {}, {}};

    for (int k0 = 0; k0 < K; k0 += 32) {
        // ---- B fragment: 16 lane-coalesced b32 loads, reused 4x ----
        v16bf b;
        const float* wp = Wcol + (size_t)(k0 + half * 8) * HID;
#pragma unroll
        for (int e = 0; e < 16; ++e) b[e] = (__bf16)wp[(size_t)(e < 8 ? e : e + 8) * HID];

        // ---- 4 A fragments (4x global_load_b128 each) + 4 WMMAs ----
        const int kA0 = k0 + half * 8;
#pragma unroll
        for (int t = 0; t < 4; ++t) {
            const float* ap = Arow + (size_t)(t * 16) * lda + kA0;
            const float4 lo0 = *(const float4*)(ap);
            const float4 lo1 = *(const float4*)(ap + 4);
            const float4 hi0 = *(const float4*)(ap + 16);
            const float4 hi1 = *(const float4*)(ap + 20);
            v16bf a;
            a[0]  = (__bf16)lo0.x; a[1]  = (__bf16)lo0.y;
            a[2]  = (__bf16)lo0.z; a[3]  = (__bf16)lo0.w;
            a[4]  = (__bf16)lo1.x; a[5]  = (__bf16)lo1.y;
            a[6]  = (__bf16)lo1.z; a[7]  = (__bf16)lo1.w;
            a[8]  = (__bf16)hi0.x; a[9]  = (__bf16)hi0.y;
            a[10] = (__bf16)hi0.z; a[11] = (__bf16)hi0.w;
            a[12] = (__bf16)hi1.x; a[13] = (__bf16)hi1.y;
            a[14] = (__bf16)hi1.z; a[15] = (__bf16)hi1.w;
            acc[t] = __builtin_amdgcn_wmma_f32_16x16x32_bf16(
                false, a, false, b, (short)0, acc[t], false, false);
        }
    }

    const float bb = bias[ncol];
#pragma unroll
    for (int t = 0; t < 4; ++t) {
#pragma unroll
        for (int r = 0; r < 8; ++r) {
            const int row = m0 + t * 16 + r + 8 * half;
            if (row < M)
                C[(size_t)row * HID + ncol] = fmaxf(acc[t][r] + bb, 0.0f);
        }
    }
}

// ---------------------------------------------------------------------------
// Column sum (block-partial + atomic combine). ld = row stride of X.
// ---------------------------------------------------------------------------
__global__ void colsum_kernel(const float* __restrict__ X, int ld,
                              float* __restrict__ sum,
                              int M, int D, int rowsPerBlock) {
    const int col = threadIdx.x;
    if (col >= D) return;
    int r0 = blockIdx.x * rowsPerBlock;
    int r1 = r0 + rowsPerBlock; if (r1 > M) r1 = M;
    float s = 0.0f;
    for (int r = r0; r < r1; ++r) s += X[(size_t)r * ld + col];
    atomicAdd(&sum[col], s);
}

// ---------------------------------------------------------------------------
// Tiny vec-mat:  out[j] = relu(sum_k v[k]*W[k*HID+j] + b[j]), one block of 256
// ---------------------------------------------------------------------------
__global__ void vecmat_relu_kernel(const float* __restrict__ v,
                                   const float* __restrict__ W,
                                   const float* __restrict__ b,
                                   float* __restrict__ out, int K) {
    const int j = threadIdx.x;
    float s = b[j];
    for (int k = 0; k < K; ++k) s += v[k] * W[(size_t)k * HID + j];
    out[j] = fmaxf(s, 0.0f);
}

__global__ void add3_kernel(const float* a, const float* b, const float* c,
                            float* out) {
    const int j = threadIdx.x;
    out[j] = a[j] + b[j] + c[j];
}

// out[i] += g[i % 256]  (HID == 256, power of two)
__global__ void bcast_add_kernel(float* __restrict__ out,
                                 const float* __restrict__ g, int n) {
    int i = blockIdx.x * blockDim.x + threadIdx.x;
    const int stride = gridDim.x * blockDim.x;
    for (; i < n; i += stride) out[i] += g[i & (HID - 1)];
}

// ---------------------------------------------------------------------------
// Orchestration
// ---------------------------------------------------------------------------
extern "C" void kernel_launch(void* const* d_in, const int* in_sizes, int n_in,
                              void* d_out, int out_size, void* d_ws, size_t ws_size,
                              hipStream_t stream) {
    (void)in_sizes; (void)n_in; (void)out_size; (void)ws_size;

    const float*     feat  = (const float*)d_in[0];
    const long long* ei    = (const long long*)d_in[1];   // int64 [2, N_EDGES]
    const float* g0_w = (const float*)d_in[2];
    const float* g0_b = (const float*)d_in[3];
    const float* g1_w = (const float*)d_in[4];
    const float* g1_b = (const float*)d_in[5];
    const float* g2_w = (const float*)d_in[6];
    const float* g2_b = (const float*)d_in[7];
    const float* gl_w = (const float*)d_in[8];
    const float* gl_b = (const float*)d_in[9];
    const float* m1a_w = (const float*)d_in[10];
    const float* m1a_b = (const float*)d_in[11];
    const float* m1b_w = (const float*)d_in[12];
    const float* m1b_b = (const float*)d_in[13];
    const float* m2a_w = (const float*)d_in[14];
    const float* m2a_b = (const float*)d_in[15];
    const float* m2b_w = (const float*)d_in[16];
    const float* m2b_b = (const float*)d_in[17];

    float* out = (float*)d_out;                      // [N_NODES, HID]

    // Workspace: 3 big buffers [MPAD x 256] + padded weight + small vectors
    const size_t BIG = (size_t)MPAD * HID;           // 12.81M floats
    float* B0 = (float*)d_ws;                        // agg (layer1: stride K1PAD)
    float* B1 = B0 + BIG;                            // h1 / h2
    float* B2 = B1 + BIG;                            // node1
    float* WP = B2 + BIG;                            // m1a_w padded [K1PAD x HID]
    float* SV = WP + (size_t)K1PAD * HID;            // small vectors, 8 x 256
    float* feat_sum  = SV + 0 * 256;
    float* node1_sum = SV + 1 * 256;
    float* node2_sum = SV + 2 * 256;
    float* g_emb0    = SV + 3 * 256;
    float* g_emb1    = SV + 4 * 256;
    float* g_emb2    = SV + 5 * 256;
    float* g_sum     = SV + 6 * 256;
    float* g_last    = SV + 7 * 256;

    const int gemmBlocks = (MPAD / 64) * (HID / 16) / 8;   // 782*16/8 = 1564

    // 0) zero accumulators; pad feat and m1a_w
    zero_kernel<<<8, 256, 0, stream>>>(SV, 8 * 256);
    pad_copy_feat<<<(N_NODES * K1PAD + 255) / 256, 256, 0, stream>>>(feat, B0,
                                                                     N_NODES);
    pad_weight<<<(K1PAD * HID + 255) / 256, 256, 0, stream>>>(m1a_w, WP);

    // ---- Layer 1: agg1 = feat + segment_sum(feat[src] -> dst) ----
    scatter_add_kernel<<<16384, 192, 0, stream>>>(feat, IN_DIM, ei, B0, K1PAD,
                                                  IN_DIM, N_EDGES);

    // h1 = relu(agg1 @ m1a + b);  node1 = relu(h1 @ m1b + b)
    gemm_bias_relu_wmma<<<gemmBlocks, 256, 0, stream>>>(B0, K1PAD, WP, m1a_b,
                                                        B1, N_NODES, K1PAD);
    gemm_bias_relu_wmma<<<gemmBlocks, 256, 0, stream>>>(B1, HID, m1b_w, m1b_b,
                                                        B2, N_NODES, HID);

    // column sums of feat and node1
    colsum_kernel<<<(N_NODES + 511) / 512, 192, 0, stream>>>(feat, IN_DIM,
                                                             feat_sum, N_NODES,
                                                             IN_DIM, 512);
    colsum_kernel<<<(N_NODES + 511) / 512, 256, 0, stream>>>(B2, HID, node1_sum,
                                                             N_NODES, HID, 512);

    // ---- Layer 2: agg2 = node1 + segment_sum(node1[src] -> dst) ----
    hipMemcpyAsync(B0, B2, (size_t)N_NODES * HID * sizeof(float),
                   hipMemcpyDeviceToDevice, stream);
    scatter_add_kernel<<<16384, 256, 0, stream>>>(B2, HID, ei, B0, HID,
                                                  HID, N_EDGES);

    // h2 = relu(agg2 @ m2a + b);  node2 = relu(h2 @ m2b + b) -> d_out
    gemm_bias_relu_wmma<<<gemmBlocks, 256, 0, stream>>>(B0, HID, m2a_w, m2a_b,
                                                        B1, N_NODES, HID);
    gemm_bias_relu_wmma<<<gemmBlocks, 256, 0, stream>>>(B1, HID, m2b_w, m2b_b,
                                                        out, N_NODES, HID);

    colsum_kernel<<<(N_NODES + 511) / 512, 256, 0, stream>>>(out, HID,
                                                             node2_sum, N_NODES,
                                                             HID, 512);

    // ---- Graph embedding chain (tiny) ----
    vecmat_relu_kernel<<<1, 256, 0, stream>>>(feat_sum,  g0_w, g0_b, g_emb0, IN_DIM);
    vecmat_relu_kernel<<<1, 256, 0, stream>>>(node1_sum, g1_w, g1_b, g_emb1, HID);
    vecmat_relu_kernel<<<1, 256, 0, stream>>>(node2_sum, g2_w, g2_b, g_emb2, HID);
    add3_kernel<<<1, 256, 0, stream>>>(g_emb0, g_emb1, g_emb2, g_sum);
    vecmat_relu_kernel<<<1, 256, 0, stream>>>(g_sum, gl_w, gl_b, g_last, HID);

    // ---- out = node2 + g_last (broadcast) ----
    bcast_add_kernel<<<(N_NODES * HID + 255) / 256, 256, 0, stream>>>(
        out, g_last, N_NODES * HID);
}
